// MultiHeadAttention4SimpleKT_21320217657940
// MI455X (gfx1250) — matrix operand order repaired and checked by reading, not compiled, not run on hardware
//
#include <hip/hip_runtime.h>
#include <hip/hip_bf16.h>

// Problem constants (from reference): B=16, S=1024, D=512, H=8, DH=64
#define SLEN   1024
#define DMODEL 512
#define NHEAD  8
#define DHEAD  64
#define BATCH  16
#define MROWS  (BATCH * SLEN)   // 16384

typedef __attribute__((ext_vector_type(16))) __bf16 v16bf;
typedef __attribute__((ext_vector_type(8)))  __bf16 bf16x8;
typedef __attribute__((ext_vector_type(8)))  float  v8f;
typedef __attribute__((ext_vector_type(4)))  float  f32x4;

static __device__ __forceinline__ v16bf pack16(bf16x8 lo, bf16x8 hi) {
  return __builtin_shufflevector(lo, hi, 0,1,2,3,4,5,6,7,8,9,10,11,12,13,14,15);
}

static __device__ __forceinline__ bf16x8 cvt_bf8(f32x4 a, f32x4 b) {
  bf16x8 r;
  r[0] = (__bf16)a[0]; r[1] = (__bf16)a[1]; r[2] = (__bf16)a[2]; r[3] = (__bf16)a[3];
  r[4] = (__bf16)b[0]; r[5] = (__bf16)b[1]; r[6] = (__bf16)b[2]; r[7] = (__bf16)b[3];
  return r;
}

// A-fragment (16x32 bf16, M x K). ISA layout: lane m=lane&15 holds
// K = {8*hi..8*hi+7} in elems 0-7 and K = {16+8*hi..23+8*hi} in elems 8-15.
static __device__ __forceinline__ v16bf load_a_f32(const float* row, int kb, int hi) {
  const float* p0 = row + kb + 8 * hi;
  const float* p1 = row + kb + 16 + 8 * hi;
  bf16x8 lo = cvt_bf8(*(const f32x4*)p0, *(const f32x4*)(p0 + 4));
  bf16x8 hi8 = cvt_bf8(*(const f32x4*)p1, *(const f32x4*)(p1 + 4));
  return pack16(lo, hi8);
}
static __device__ __forceinline__ v16bf load_a_bf16(const __bf16* row, int kb, int hi) {
  const __bf16* p0 = row + kb + 8 * hi;
  const __bf16* p1 = row + kb + 16 + 8 * hi;
  return pack16(*(const bf16x8*)p0, *(const bf16x8*)p1);
}
// B-fragment (32x16 bf16, K x N). ISA layout: lane n=lane&15 holds
// K = {8*hi..8*hi+7} in elems 0-7 and K = {16+8*hi..23+8*hi} in elems 8-15. colrow points at element K=0 of column n.
static __device__ __forceinline__ v16bf load_b_f32(const float* colrow, int kb, int hi) {
  const float* p0 = colrow + kb + 8 * hi;
  const float* p1 = colrow + kb + 16 + 8 * hi;
  bf16x8 lo = cvt_bf8(*(const f32x4*)p0, *(const f32x4*)(p0 + 4));
  bf16x8 hi8 = cvt_bf8(*(const f32x4*)p1, *(const f32x4*)(p1 + 4));
  return pack16(lo, hi8);
}
static __device__ __forceinline__ v16bf load_b_bf16(const __bf16* colrow, int kb, int hi) {
  const __bf16* p0 = colrow + kb + 8 * hi;
  const __bf16* p1 = colrow + kb + 16 + 8 * hi;
  return pack16(*(const bf16x8*)p0, *(const bf16x8*)p1);
}

#define WMMA_BF16(A, B, C) \
  __builtin_amdgcn_wmma_f32_16x16x32_bf16(false, (A), false, (B), (short)0, (C), false, false)

// Output layouts for the projection GEMM
#define LAY_PLAIN   0   // [MROWS, DMODEL]
#define LAY_HEAD    1   // [B, H, S, DH]   (Q, K operand layouts)
#define LAY_HEAD_T  2   // [B, H, DH, S]   (V: key-contiguous for PV B-fragments)

static __device__ __forceinline__ size_t proj_idx(int LAYOUT, int row, int n) {
  if (LAYOUT == LAY_HEAD) {
    const int bb = row >> 10, s = row & (SLEN - 1);
    const int h = n >> 6, d = n & (DHEAD - 1);
    return (((size_t)(bb * NHEAD + h) * SLEN) + s) * DHEAD + d;
  } else if (LAYOUT == LAY_HEAD_T) {
    const int bb = row >> 10, s = row & (SLEN - 1);
    const int h = n >> 6, d = n & (DHEAD - 1);
    return (((size_t)(bb * NHEAD + h) * DHEAD) + d) * SLEN + s;
  }
  return (size_t)row * DMODEL + n;
}

// ------------- Projection: Y = X @ W^T + bias, fp32 -> bf16 ----------------
// 16x64 output tile per wave: one shared A fragment feeds 4 WMMAs per K-step.
template <int LAYOUT>
__global__ void __launch_bounds__(256)
proj_f32_bf16(const float* __restrict__ X, const float* __restrict__ W,
              const float* __restrict__ bias, __bf16* __restrict__ Y) {
  const int lane = threadIdx.x & 31;
  const int wv = threadIdx.x >> 5;
  const int w = blockIdx.x * 8 + wv;
  const int tn = w & 7;     // 8 N-tiles of 64 cols
  const int tm = w >> 3;    // 1024 M-tiles (all 8 waves of a block share tm)
  const int m = lane & 15, hi = lane >> 4;
  const float* xrow = X + (size_t)(tm * 16 + m) * DMODEL;
  const float* wrow0 = W + (size_t)(tn * 64 + m) * DMODEL;  // col groups g*16
  v8f acc[4] = {{}, {}, {}, {}};
#pragma unroll 1
  for (int kb = 0; kb < DMODEL; kb += 32) {
    __builtin_prefetch(xrow + kb + 64, 0, 1);
    const v16bf a = load_a_f32(xrow, kb, hi);
#pragma unroll
    for (int g = 0; g < 4; ++g) {
      const v16bf b = load_b_f32(wrow0 + (size_t)g * 16 * DMODEL, kb, hi);
      acc[g] = WMMA_BF16(a, b, acc[g]);
    }
  }
#pragma unroll
  for (int g = 0; g < 4; ++g) {
    const int n = tn * 64 + g * 16 + m;
    const float bv = bias[n];
#pragma unroll
    for (int r = 0; r < 8; ++r) {
      const int row = tm * 16 + r + 8 * hi;   // C layout: M = vgpr + 8*(lane>=16)
      Y[proj_idx(LAYOUT, row, n)] = (__bf16)(acc[g][r] + bv);
    }
  }
}

// ------------- Output projection: bf16 -> fp32 (16x64 tile per wave) -------
__global__ void __launch_bounds__(256)
proj_bf16_f32(const __bf16* __restrict__ X, const float* __restrict__ W,
              const float* __restrict__ bias, float* __restrict__ Y) {
  const int lane = threadIdx.x & 31;
  const int wv = threadIdx.x >> 5;
  const int w = blockIdx.x * 8 + wv;
  const int tn = w & 7;
  const int tm = w >> 3;
  const int m = lane & 15, hi = lane >> 4;
  const __bf16* xrow = X + (size_t)(tm * 16 + m) * DMODEL;
  const float* wrow0 = W + (size_t)(tn * 64 + m) * DMODEL;
  v8f acc[4] = {{}, {}, {}, {}};
#pragma unroll 1
  for (int kb = 0; kb < DMODEL; kb += 32) {
    const v16bf a = load_a_bf16(xrow, kb, hi);
#pragma unroll
    for (int g = 0; g < 4; ++g) {
      const v16bf b = load_b_f32(wrow0 + (size_t)g * 16 * DMODEL, kb, hi);
      acc[g] = WMMA_BF16(a, b, acc[g]);
    }
  }
#pragma unroll
  for (int g = 0; g < 4; ++g) {
    const int n = tn * 64 + g * 16 + m;
    const float bv = bias[n];
#pragma unroll
    for (int r = 0; r < 8; ++r) {
      const int row = tm * 16 + r + 8 * hi;
      Y[(size_t)row * DMODEL + n] = acc[g][r] + bv;
    }
  }
}

// ---------------- Fused causal flash-attention -----------------------------
// One wave per (b,h, 16-query tile). Streams 32-key blocks, online softmax,
// P transposed C-layout -> A-layout through LDS, PV via WMMA.
// vh is stored [B,H,DH,S] so V B-fragments are two b128 loads per fragment.
__global__ void __launch_bounds__(128)
attn_fused(const __bf16* __restrict__ qh, const __bf16* __restrict__ kh,
           const __bf16* __restrict__ vh, __bf16* __restrict__ out,
           const int* __restrict__ zero_pad) {
  __shared__ __align__(16) __bf16 pbuf[4][16 * 32];
  const int lane = threadIdx.x & 31;
  const int wv = threadIdx.x >> 5;
  const int w = blockIdx.x * 4 + wv;
  const int qt = w & 63;       // 64 query tiles per (b,h)
  const int bh = w >> 6;       // 0..127
  const int q0 = qt * 16;
  const int m = lane & 15, hi = lane >> 4;
  const __bf16* Qb = qh + (size_t)bh * SLEN * DHEAD;
  const __bf16* Kb = kh + (size_t)bh * SLEN * DHEAD;
  const __bf16* Vb = vh + (size_t)bh * DHEAD * SLEN;  // [DH, S]

  // Q A-fragments: 16 queries x DH=64 (two K=32 chunks)
  const __bf16* qrow = Qb + (size_t)(q0 + m) * DHEAD;
  const v16bf qa0 = load_a_bf16(qrow, 0, hi);
  const v16bf qa1 = load_a_bf16(qrow, 32, hi);

  v8f acc[4] = {{}, {}, {}, {}};   // 16 x 64 output accumulator (4 N-frags)
  float mst[8], lst[8];
#pragma unroll
  for (int r = 0; r < 8; ++r) { mst[r] = -3.0e38f; lst[r] = 0.0f; }

  __bf16* lp = &pbuf[wv][0];
  const int nkb = (q0 + 16 + 31) >> 5;  // 32-key causal blocks
  for (int kb = 0; kb < nkb; ++kb) {
    const int key0 = kb * 32;
    // scores = Q . K^T : two 16-col fragments, K-dim = DH (2 x 32)
    v8f sc0 = {}, sc1 = {};
    {
      const __bf16* kr0 = Kb + (size_t)(key0 + m) * DHEAD;       // cols 0-15
      const __bf16* kr1 = Kb + (size_t)(key0 + 16 + m) * DHEAD;  // cols 16-31
      sc0 = WMMA_BF16(qa0, load_b_bf16(kr0, 0, hi), sc0);
      sc0 = WMMA_BF16(qa1, load_b_bf16(kr0, 32, hi), sc0);
      sc1 = WMMA_BF16(qa0, load_b_bf16(kr1, 0, hi), sc1);
      sc1 = WMMA_BF16(qa1, load_b_bf16(kr1, 32, hi), sc1);
    }
    // scale, causal mask, online softmax (C layout: col = lane&15, row = r+8*hi)
#pragma unroll
    for (int r = 0; r < 8; ++r) {
      const int qrow_g = q0 + r + 8 * hi;
      float s0 = sc0[r] * 0.125f;               // 1/sqrt(64)
      float s1 = sc1[r] * 0.125f;
      if (key0 + m > qrow_g)      s0 = -3.0e38f;
      if (key0 + 16 + m > qrow_g) s1 = -3.0e38f;
      float mx = fmaxf(s0, s1);
#pragma unroll
      for (int d = 1; d < 16; d <<= 1) mx = fmaxf(mx, __shfl_xor(mx, d, 32));
      const float mnew = fmaxf(mst[r], mx);
      const float alpha = __expf(mst[r] - mnew);
      const float p0 = __expf(s0 - mnew);
      const float p1 = __expf(s1 - mnew);
      float ls = p0 + p1;
#pragma unroll
      for (int d = 1; d < 16; d <<= 1) ls += __shfl_xor(ls, d, 32);
      lst[r] = lst[r] * alpha + ls;
      mst[r] = mnew;
#pragma unroll
      for (int nf = 0; nf < 4; ++nf) acc[nf][r] *= alpha;
      const int row = r + 8 * hi;
      lp[row * 32 + m]      = (__bf16)p0;        // transpose staging
      lp[row * 32 + 16 + m] = (__bf16)p1;
    }
    asm volatile("s_wait_dscnt 0" ::: "memory");  // same-wave LDS RAW
    // P as A-fragment (16 queries x 32 keys)
    const __bf16* rp = lp + m * 32 + 8 * hi;
    const v16bf pa = pack16(*(const bf16x8*)rp, *(const bf16x8*)(rp + 16));
    // V B-fragments from [DH,S] layout: lane n = d = nf*16 + m,
    // elems 0-7 = keys key0+8*hi.., elems 8-15 = keys key0+16+8*hi.. -> two b128 loads.
#pragma unroll
    for (int nf = 0; nf < 4; ++nf) {
      const __bf16* vcol = Vb + (size_t)(nf * 16 + m) * SLEN;
      acc[nf] = WMMA_BF16(pa, load_b_bf16(vcol, key0, hi), acc[nf]);
    }
  }

  // epilogue: normalize, zero_pad row 0, store concat layout [B,S,D]
  const int zp = zero_pad[0];
  const int bb = bh >> 3, h = bh & 7;
#pragma unroll
  for (int r = 0; r < 8; ++r) {
    const int qrow_g = q0 + r + 8 * hi;
    float invl = lst[r] > 0.0f ? 1.0f / lst[r] : 0.0f;
    if (zp && qrow_g == 0) invl = 0.0f;
#pragma unroll
    for (int nf = 0; nf < 4; ++nf) {
      const int d = nf * 16 + m;
      const size_t idx = ((size_t)bb * SLEN + qrow_g) * DMODEL + h * DHEAD + d;
      out[idx] = (__bf16)(acc[nf][r] * invl);
    }
  }
}

extern "C" void kernel_launch(void* const* d_in, const int* in_sizes, int n_in,
                              void* d_out, int out_size, void* d_ws, size_t ws_size,
                              hipStream_t stream) {
  (void)in_sizes; (void)n_in; (void)out_size; (void)ws_size;
  const float* q  = (const float*)d_in[0];
  const float* k  = (const float*)d_in[1];
  const float* v  = (const float*)d_in[2];
  // d_in[3] = causal mask (implicit in kernel)
  const float* Wk = (const float*)d_in[4];
  const float* bk = (const float*)d_in[5];
  const float* Wv = (const float*)d_in[6];
  const float* bv = (const float*)d_in[7];
  const float* Wo = (const float*)d_in[8];
  const float* bo = (const float*)d_in[9];
  const int*   zp = (const int*)d_in[10];

  const size_t elems = (size_t)BATCH * SLEN * DMODEL;  // 8M bf16 = 16 MB each
  __bf16* qh = (__bf16*)d_ws;
  __bf16* kh = qh + elems;
  __bf16* vh = kh + elems;
  __bf16* ao = vh + elems;

  const int proj_blocks = (MROWS / 16) * (DMODEL / 64) / 8;     // 1024
  const int attn_blocks = (BATCH * NHEAD) * (SLEN / 16) / 4;    // 2048

  proj_f32_bf16<LAY_HEAD><<<proj_blocks, 256, 0, stream>>>(q, Wk, bk, qh);
  proj_f32_bf16<LAY_HEAD><<<proj_blocks, 256, 0, stream>>>(k, Wk, bk, kh);
  proj_f32_bf16<LAY_HEAD_T><<<proj_blocks, 256, 0, stream>>>(v, Wv, bv, vh);
  attn_fused<<<attn_blocks, 128, 0, stream>>>(qh, kh, vh, ao, zp);
  proj_bf16_f32<<<proj_blocks, 256, 0, stream>>>(ao, Wo, bo, (float*)d_out);
}
